// TransformerWithDSA_75316546503195
// MI455X (gfx1250) — matrix operand order repaired and checked by reading, not compiled, not run on hardware
//
#include <hip/hip_runtime.h>
#include <hip/hip_bf16.h>

// ---------- types ----------
typedef __bf16 bf16_t;
typedef __attribute__((ext_vector_type(16))) __bf16 v16bf;
typedef __attribute__((ext_vector_type(8)))  __bf16 v8bf;
typedef __attribute__((ext_vector_type(8)))  float   v8f;

struct frag16 { v8bf lo, hi; };

__device__ __forceinline__ v16bf frag_cast(const frag16 &f) {
  return __builtin_bit_cast(v16bf, f);
}

__device__ __forceinline__ bf16_t f2bf(float f) {
  union { float f; unsigned u; } x; x.f = f;
  unsigned r = x.u + 0x7fffu + ((x.u >> 16) & 1u);   // round-to-nearest-even
  unsigned short h = (unsigned short)(r >> 16);
  return __builtin_bit_cast(bf16_t, h);
}

__device__ __forceinline__ unsigned pack2(bf16_t a, bf16_t b) {
  return (unsigned)__builtin_bit_cast(unsigned short, a)
       | ((unsigned)__builtin_bit_cast(unsigned short, b) << 16);
}

__device__ __forceinline__ v8f wmma_bf16(v16bf a, v16bf b, v8f c) {
  return __builtin_amdgcn_wmma_f32_16x16x32_bf16(false, a, false, b, (short)0, c, false, false);
}

// ---------- gfx1250 async global->LDS (ASYNCcnt-tracked, no VGPR round-trip) --
// GVS form: mem = SADDR(u64) + VADDR(i32) + inst_offset ; LDS dest = VDST + inst_offset
__device__ __forceinline__ void async_to_lds_64B(void* lds, const void* gbase,
                                                 unsigned goff) {
  unsigned l = (unsigned)(unsigned long long)lds;     // low 32 bits = LDS offset
  asm volatile("global_load_async_to_lds_b128 %0, %1, %2\n\t"
               "global_load_async_to_lds_b128 %0, %1, %2 offset:16\n\t"
               "global_load_async_to_lds_b128 %0, %1, %2 offset:32\n\t"
               "global_load_async_to_lds_b128 %0, %1, %2 offset:48"
               :: "v"(l), "v"(goff), "s"(gbase) : "memory");
}
__device__ __forceinline__ void async_to_lds_32B(void* lds, const void* gbase,
                                                 unsigned goff) {
  unsigned l = (unsigned)(unsigned long long)lds;
  asm volatile("global_load_async_to_lds_b128 %0, %1, %2\n\t"
               "global_load_async_to_lds_b128 %0, %1, %2 offset:16"
               :: "v"(l), "v"(goff), "s"(gbase) : "memory");
}
__device__ __forceinline__ void wait_async0() {
  asm volatile("s_wait_asynccnt 0x0" ::: "memory");
}

#define NT   16384   // B*S tokens
#define EMB  512
#define FFN  2048
#define SEQ  1024
#define NH   8
#define DH   64

// =====================================================================
// bf16 WMMA GEMM. BM=128, BN=128, BK=64; 256 threads = 8 waves (4Mx2N),
// each wave 2x4 WMMA tiles; 16 WMMAs per K iteration.
// Ping-pong LDS double buffer, ONE barrier per tile:
//   A tile : global_load_async_to_lds_b128 (ASYNCcnt)
//   B tile : register-staged transpose (packed-pair b32 LDS stores)
// =====================================================================
template<bool PE, bool RELU, bool OUTF, bool OUTB>
__global__ __launch_bounds__(256)
void k_gemm(const bf16_t* __restrict__ A, const bf16_t* __restrict__ B,
            const float* __restrict__ bias, const float* __restrict__ pe,
            float* __restrict__ outf, bf16_t* __restrict__ outb,
            int N, int K)
{
  __shared__ __align__(16) bf16_t sA[2][128][64];   // [m][k] 2x16KB
  __shared__ __align__(16) bf16_t sB[2][128][64];   // [n][k] 2x16KB (transposed)

  const int tid  = threadIdx.x;
  const int lane = tid & 31;
  const int wave = tid >> 5;
  const int hi   = lane >> 4;
  const int l16  = lane & 15;
  const int bm   = blockIdx.x * 128;
  const int bn   = blockIdx.y * 128;
  const int wm   = (wave >> 1) * 32;
  const int wn   = (wave & 1) * 64;

  // staging maps
  const int arow = tid >> 1;            // A: 128 rows x 64 cols, 64B/thread
  const int acol = (tid & 1) * 32;
  const int bk2  = (tid & 31) * 2;      // B: paired K rows -> packed b32 stores
  const int bns  = (tid >> 5) * 16;     // 8 groups x 16 n = 128

  v8bf rb0[2], rb1[2];

  auto asyncA = [&](int k0, int p) {
    const unsigned go = (unsigned)((((size_t)(bm + arow)) * K + k0 + acol) * 2);
    async_to_lds_64B(&sA[p][arow][acol], A, go);
  };
  auto loadB = [&](int k0) {
    const bf16_t* gb = B + (size_t)(k0 + bk2) * N + bn + bns;
    rb0[0] = *(const v8bf*)(gb);
    rb0[1] = *(const v8bf*)(gb + 8);
    rb1[0] = *(const v8bf*)(gb + N);
    rb1[1] = *(const v8bf*)(gb + N + 8);
  };
  auto storeB = [&](int p) {
    #pragma unroll
    for (int v = 0; v < 2; ++v)
      #pragma unroll
      for (int j = 0; j < 8; ++j)
        *(unsigned*)&sB[p][bns + v * 8 + j][bk2] = pack2(rb0[v][j], rb1[v][j]);
  };

  v8f acc[2][4];
  #pragma unroll
  for (int i = 0; i < 2; ++i)
    #pragma unroll
    for (int j = 0; j < 4; ++j)
      #pragma unroll
      for (int e = 0; e < 8; ++e) acc[i][j][e] = 0.0f;

  // prologue: tile 0 in flight
  asyncA(0, 0);
  loadB(0);
  storeB(0);

  const int ntile = K >> 6;
  for (int it = 0; it < ntile; ++it) {
    const int p = it & 1;
    wait_async0();                 // my async stores into sA[p] done
    __syncthreads();               // all waves' ds-stores visible / done reading p^1
    if (it + 1 < ntile) {          // next tile into other buffer, overlaps WMMA
      asyncA((it + 1) * 64, p ^ 1);
      loadB((it + 1) * 64);
      storeB(p ^ 1);
      if (it + 2 < ntile)          // gfx1250 prefetch for tile after next
        __builtin_prefetch(B + (size_t)((it + 2) * 64 + bk2) * N + bn + bns, 0, 0);
    }
    #pragma unroll
    for (int ks = 0; ks < 2; ++ks) {
      v16bf av[2], bv[4];
      #pragma unroll
      for (int i = 0; i < 2; ++i) {
        frag16 f;
        const int r = wm + i * 16 + l16;
        f.lo = *(const v8bf*)&sA[p][r][ks * 32 + 8 * hi];
        f.hi = *(const v8bf*)&sA[p][r][ks * 32 + 16 + 8 * hi];
        av[i] = frag_cast(f);
      }
      #pragma unroll
      for (int j = 0; j < 4; ++j) {
        frag16 f;
        const int n = wn + j * 16 + l16;
        f.lo = *(const v8bf*)&sB[p][n][ks * 32 + 8 * hi];
        f.hi = *(const v8bf*)&sB[p][n][ks * 32 + 16 + 8 * hi];
        bv[j] = frag_cast(f);
      }
      #pragma unroll
      for (int i = 0; i < 2; ++i)
        #pragma unroll
        for (int j = 0; j < 4; ++j)
          acc[i][j] = wmma_bf16(av[i], bv[j], acc[i][j]);
    }
  }

  // --- specialized epilogue ---
  #pragma unroll
  for (int i = 0; i < 2; ++i) {
    #pragma unroll
    for (int j = 0; j < 4; ++j) {
      const int gn = bn + wn + j * 16 + l16;
      const float bvv = bias[gn];
      #pragma unroll
      for (int r = 0; r < 8; ++r) {
        const int gm = bm + wm + i * 16 + r + 8 * hi;
        float v = acc[i][j][r] + bvv;
        if (PE)   v += pe[(gm >> 10) * EMB + gn];   // pe[B,1,E] bcast over seq
        if (RELU) v = v > 0.0f ? v : 0.0f;
        const size_t idx = (size_t)gm * N + gn;
        if (OUTF) outf[idx] = v;
        if (OUTB) outb[idx] = f2bf(v);
      }
    }
  }
}

// =====================================================================
// Flash attention: per block one (b,head), 128 query rows (8 waves x 16).
// Q fragments resident in registers; K staged via async-to-LDS ping-pong,
// V^T register-transposed into ping-pong LDS; one barrier per key block;
// online softmax with shfl_xor row reductions over 16-lane C groups.
// =====================================================================
__global__ __launch_bounds__(256)
void k_attn(const bf16_t* __restrict__ Q, const bf16_t* __restrict__ Kt,
            const bf16_t* __restrict__ V, bf16_t* __restrict__ O)
{
  __shared__ __align__(16) bf16_t sK[2][64][64];     // [key][d]  2x8KB
  __shared__ __align__(16) bf16_t sVt[2][64][64];    // [d][key]  2x8KB
  __shared__ __align__(16) bf16_t sP[8][16][64];     // per-wave P 16KB

  const int tid  = threadIdx.x;
  const int lane = tid & 31;
  const int wave = tid >> 5;
  const int hi   = lane >> 4;
  const int l16  = lane & 15;
  const int bh   = blockIdx.x;                 // 0..127
  const int b    = bh >> 3;
  const int head = bh & 7;
  const int q0   = blockIdx.y * 128 + wave * 16;
  const float scale = 0.125f;                  // 1/sqrt(64)
  const size_t base = (size_t)b * SEQ * EMB + head * DH;

  // staging maps
  const int krow = tid >> 2;                   // K: 64 rows x 64 d, 32B/thread
  const int kdc  = (tid & 3) * 16;
  const int vk2  = (tid & 31) * 2;             // V: paired key rows
  const int vdc  = (tid >> 5) * 8;             // 8 d per thread

  v8bf rv0, rv1;
  auto asyncK = [&](int kb, int p) {
    const unsigned go =
        (unsigned)((base + (size_t)(kb + krow) * EMB + kdc) * 2);
    async_to_lds_32B(&sK[p][krow][kdc], Kt, go);
  };
  auto loadV = [&](int kb) {
    const bf16_t* vp = V + base + (size_t)(kb + vk2) * EMB + vdc;
    rv0 = *(const v8bf*)vp;
    rv1 = *(const v8bf*)(vp + EMB);
  };
  auto storeV = [&](int p) {
    #pragma unroll
    for (int j = 0; j < 8; ++j)
      *(unsigned*)&sVt[p][vdc + j][vk2] = pack2(rv0[j], rv1[j]);
  };

  // Q fragments (A-layout, straight from global, stay resident)
  v16bf qv[2];
  {
    const bf16_t* qp = Q + base + (size_t)(q0 + l16) * EMB;
    frag16 f;
    f.lo = *(const v8bf*)(qp + 8 * hi);
    f.hi = *(const v8bf*)(qp + 16 + 8 * hi);
    qv[0] = frag_cast(f);
    f.lo = *(const v8bf*)(qp + 32 + 8 * hi);
    f.hi = *(const v8bf*)(qp + 32 + 16 + 8 * hi);
    qv[1] = frag_cast(f);
  }

  float mrow[8], lrow[8];
  v8f o[4];
  #pragma unroll
  for (int r = 0; r < 8; ++r) { mrow[r] = -1e30f; lrow[r] = 0.0f; }
  #pragma unroll
  for (int d = 0; d < 4; ++d)
    #pragma unroll
    for (int e = 0; e < 8; ++e) o[d][e] = 0.0f;

  // prologue: block 0 in flight
  asyncK(0, 0);
  loadV(0);
  storeV(0);

  for (int it = 0; it < SEQ / 64; ++it) {
    const int p = it & 1;
    wait_async0();
    __syncthreads();
    if (it + 1 < SEQ / 64) {       // next K/V block overlaps compute
      asyncK((it + 1) * 64, p ^ 1);
      loadV((it + 1) * 64);
      storeV(p ^ 1);
    }

    // ---- S = Q K^T (4 key tiles x 2 d-steps) ----
    v8f s[4];
    #pragma unroll
    for (int j = 0; j < 4; ++j) {
      #pragma unroll
      for (int e = 0; e < 8; ++e) s[j][e] = 0.0f;
      const int n = j * 16 + l16;
      frag16 f;
      f.lo = *(const v8bf*)&sK[p][n][8 * hi];
      f.hi = *(const v8bf*)&sK[p][n][16 + 8 * hi];
      s[j] = wmma_bf16(qv[0], frag_cast(f), s[j]);
      f.lo = *(const v8bf*)&sK[p][n][32 + 8 * hi];
      f.hi = *(const v8bf*)&sK[p][n][32 + 16 + 8 * hi];
      s[j] = wmma_bf16(qv[1], frag_cast(f), s[j]);
    }

    // ---- online softmax (row groups of 16 lanes) ----
    #pragma unroll
    for (int r = 0; r < 8; ++r) {
      float mx = s[0][r] * scale;
      #pragma unroll
      for (int j = 1; j < 4; ++j) mx = fmaxf(mx, s[j][r] * scale);
      mx = fmaxf(mx, __shfl_xor(mx, 1, 32));
      mx = fmaxf(mx, __shfl_xor(mx, 2, 32));
      mx = fmaxf(mx, __shfl_xor(mx, 4, 32));
      mx = fmaxf(mx, __shfl_xor(mx, 8, 32));
      const float mnew  = fmaxf(mrow[r], mx);
      const float alpha = __expf(mrow[r] - mnew);
      float rs = 0.0f;
      #pragma unroll
      for (int j = 0; j < 4; ++j) {
        const float pv = __expf(s[j][r] * scale - mnew);
        s[j][r] = pv;
        rs += pv;
      }
      rs += __shfl_xor(rs, 1, 32);
      rs += __shfl_xor(rs, 2, 32);
      rs += __shfl_xor(rs, 4, 32);
      rs += __shfl_xor(rs, 8, 32);
      lrow[r] = lrow[r] * alpha + rs;
      mrow[r] = mnew;
      #pragma unroll
      for (int d = 0; d < 4; ++d) o[d][r] *= alpha;
    }

    // ---- P: C-layout -> A-layout through per-wave LDS scratch ----
    #pragma unroll
    for (int j = 0; j < 4; ++j)
      #pragma unroll
      for (int r = 0; r < 8; ++r)
        sP[wave][r + 8 * hi][j * 16 + l16] = f2bf(s[j][r]);

    v16bf pv[2];
    #pragma unroll
    for (int ks = 0; ks < 2; ++ks) {
      frag16 f;
      f.lo = *(const v8bf*)&sP[wave][l16][ks * 32 + 8 * hi];
      f.hi = *(const v8bf*)&sP[wave][l16][ks * 32 + 16 + 8 * hi];
      pv[ks] = frag_cast(f);
    }

    // ---- O += P @ V (4 d tiles x 2 key-steps) ----
    #pragma unroll
    for (int dj = 0; dj < 4; ++dj) {
      const int d = dj * 16 + l16;
      frag16 f;
      f.lo = *(const v8bf*)&sVt[p][d][8 * hi];
      f.hi = *(const v8bf*)&sVt[p][d][16 + 8 * hi];
      o[dj] = wmma_bf16(pv[0], frag_cast(f), o[dj]);
      f.lo = *(const v8bf*)&sVt[p][d][32 + 8 * hi];
      f.hi = *(const v8bf*)&sVt[p][d][32 + 16 + 8 * hi];
      o[dj] = wmma_bf16(pv[1], frag_cast(f), o[dj]);
    }
  }

  // ---- normalize + store bf16 ----
  #pragma unroll
  for (int dj = 0; dj < 4; ++dj) {
    #pragma unroll
    for (int r = 0; r < 8; ++r) {
      const float inv = 1.0f / lrow[r];
      const int row = q0 + r + 8 * hi;
      O[base + (size_t)row * EMB + dj * 16 + l16] = f2bf(o[dj][r] * inv);
    }
  }
}

// =====================================================================
// Fused residual-add + LayerNorm. One wave per row (E=512 -> 16 elems/lane).
// =====================================================================
template<bool HASR, bool OUTF, bool OUTB>
__global__ __launch_bounds__(256)
void k_ln(const float* __restrict__ X, const float* __restrict__ R,
          const float* __restrict__ g, const float* __restrict__ be,
          float* __restrict__ outf, bf16_t* __restrict__ outb)
{
  const int lane = threadIdx.x & 31;
  const int wave = threadIdx.x >> 5;
  const int row  = blockIdx.x * 8 + wave;
  const size_t rb = (size_t)row * EMB;

  float v[16];
  #pragma unroll
  for (int e = 0; e < 16; ++e) v[e] = X[rb + lane + e * 32];
  if (HASR)
    #pragma unroll
    for (int e = 0; e < 16; ++e) v[e] += R[rb + lane + e * 32];

  float s = 0.0f;
  #pragma unroll
  for (int e = 0; e < 16; ++e) s += v[e];
  s += __shfl_xor(s, 1, 32);  s += __shfl_xor(s, 2, 32);
  s += __shfl_xor(s, 4, 32);  s += __shfl_xor(s, 8, 32);
  s += __shfl_xor(s, 16, 32);
  const float mean = s * (1.0f / EMB);

  float vs = 0.0f;
  #pragma unroll
  for (int e = 0; e < 16; ++e) { const float d = v[e] - mean; vs += d * d; }
  vs += __shfl_xor(vs, 1, 32);  vs += __shfl_xor(vs, 2, 32);
  vs += __shfl_xor(vs, 4, 32);  vs += __shfl_xor(vs, 8, 32);
  vs += __shfl_xor(vs, 16, 32);
  const float inv = rsqrtf(vs * (1.0f / EMB) + 1e-5f);

  #pragma unroll
  for (int e = 0; e < 16; ++e) {
    const int c = lane + e * 32;
    const float y = (v[e] - mean) * inv * g[c] + be[c];
    if (OUTF) outf[rb + c] = y;
    if (OUTB) outb[rb + c] = f2bf(y);
  }
}

// =====================================================================
// Tiny output head: out[NT,2] = A[NT,512] @ Wout[512,2] + bout  (33 MFLOP)
// =====================================================================
__global__ __launch_bounds__(256)
void k_out(const float* __restrict__ A, const float* __restrict__ W,
           const float* __restrict__ bo, float* __restrict__ out)
{
  const int idx = blockIdx.x * blockDim.x + threadIdx.x;   // 0..32767
  const int row = idx >> 1;
  const int j   = idx & 1;
  float acc = bo[j];
  const float* ap = A + (size_t)row * EMB;
  for (int e = 0; e < EMB; ++e) acc += ap[e] * W[e * 2 + j];
  out[row * 2 + j] = acc;
}

__global__ void k_cvt(const float* __restrict__ in, bf16_t* __restrict__ out, int n)
{
  const int i = blockIdx.x * blockDim.x + threadIdx.x;
  if (i < n) out[i] = f2bf(in[i]);
}

// =====================================================================
extern "C" void kernel_launch(void* const* d_in, const int* in_sizes, int n_in,
                              void* d_out, int out_size, void* d_ws, size_t ws_size,
                              hipStream_t stream)
{
  (void)in_sizes; (void)n_in; (void)out_size; (void)ws_size;
  const float* x    = (const float*)d_in[0];
  const float* Win  = (const float*)d_in[1];
  const float* b_in = (const float*)d_in[2];
  const float* pe   = (const float*)d_in[3];
  const float* Wq   = (const float*)d_in[4];
  const float* bq   = (const float*)d_in[5];
  const float* Wk   = (const float*)d_in[6];
  const float* bk   = (const float*)d_in[7];
  const float* Wv   = (const float*)d_in[8];
  const float* bv   = (const float*)d_in[9];
  const float* Wo   = (const float*)d_in[10];
  const float* bo   = (const float*)d_in[11];
  const float* g1   = (const float*)d_in[12];
  const float* be1  = (const float*)d_in[13];
  const float* W1   = (const float*)d_in[14];
  const float* b1   = (const float*)d_in[15];
  const float* W2   = (const float*)d_in[16];
  const float* b2   = (const float*)d_in[17];
  const float* g2   = (const float*)d_in[18];
  const float* be2  = (const float*)d_in[19];
  const float* gf   = (const float*)d_in[20];
  const float* bef  = (const float*)d_in[21];
  const float* Wout = (const float*)d_in[22];
  const float* bout = (const float*)d_in[23];

  const int L = 6;
  char* wsp = (char*)d_ws;
  auto alloc = [&](size_t bytes) -> char* {
    char* p = wsp; wsp += (bytes + 255) & ~(size_t)255; return p;
  };

  bf16_t* xb   = (bf16_t*)alloc((size_t)NT * 64 * 2);
  bf16_t* Winb = (bf16_t*)alloc((size_t)64 * EMB * 2);
  bf16_t* Wqb  = (bf16_t*)alloc((size_t)L * EMB * EMB * 2);
  bf16_t* Wkb  = (bf16_t*)alloc((size_t)L * EMB * EMB * 2);
  bf16_t* Wvb  = (bf16_t*)alloc((size_t)L * EMB * EMB * 2);
  bf16_t* Wob  = (bf16_t*)alloc((size_t)L * EMB * EMB * 2);
  bf16_t* W1b  = (bf16_t*)alloc((size_t)L * EMB * FFN * 2);
  bf16_t* W2b  = (bf16_t*)alloc((size_t)L * FFN * EMB * 2);
  float*  hf   = (float*) alloc((size_t)NT * EMB * 4);
  bf16_t* hb   = (bf16_t*)alloc((size_t)NT * EMB * 2);
  bf16_t* qb   = (bf16_t*)alloc((size_t)NT * EMB * 2);
  bf16_t* kbuf = (bf16_t*)alloc((size_t)NT * EMB * 2);
  bf16_t* vbuf = (bf16_t*)alloc((size_t)NT * EMB * 2);
  bf16_t* ob   = (bf16_t*)alloc((size_t)NT * EMB * 2);
  float*  t1f  = (float*) alloc((size_t)NT * EMB * 4);
  float*  af   = (float*) alloc((size_t)NT * EMB * 4);
  bf16_t* ab   = (bf16_t*)alloc((size_t)NT * EMB * 2);
  bf16_t* f1b  = (bf16_t*)alloc((size_t)NT * FFN * 2);

  auto CVT = [&](const float* src, bf16_t* dst, int n) {
    k_cvt<<<(n + 255) / 256, 256, 0, stream>>>(src, dst, n);
  };
  CVT(x,   xb,   NT * 64);
  CVT(Win, Winb, 64 * EMB);
  CVT(Wq,  Wqb,  L * EMB * EMB);
  CVT(Wk,  Wkb,  L * EMB * EMB);
  CVT(Wv,  Wvb,  L * EMB * EMB);
  CVT(Wo,  Wob,  L * EMB * EMB);
  CVT(W1,  W1b,  L * EMB * FFN);
  CVT(W2,  W2b,  L * FFN * EMB);

  const dim3 blk(256);
  const dim3 gN512(NT / 128, EMB / 128);    // (128,4)
  const dim3 gN2048(NT / 128, FFN / 128);   // (128,16)
  const dim3 gAttn(16 * NH, SEQ / 128);     // (128,8)

  // h = x @ Win + b_in + pe   (f32 + bf16 outputs)
  k_gemm<true, false, true, true><<<gN512, blk, 0, stream>>>(
      xb, Winb, b_in, pe, hf, hb, EMB, 64);

  for (int i = 0; i < L; ++i) {
    const size_t wE = (size_t)i * EMB * EMB;
    k_gemm<false, false, false, true><<<gN512, blk, 0, stream>>>(
        hb, Wqb + wE, bq + i * EMB, nullptr, nullptr, qb, EMB, EMB);
    k_gemm<false, false, false, true><<<gN512, blk, 0, stream>>>(
        hb, Wkb + wE, bk + i * EMB, nullptr, nullptr, kbuf, EMB, EMB);
    k_gemm<false, false, false, true><<<gN512, blk, 0, stream>>>(
        hb, Wvb + wE, bv + i * EMB, nullptr, nullptr, vbuf, EMB, EMB);
    k_attn<<<gAttn, blk, 0, stream>>>(qb, kbuf, vbuf, ob);
    k_gemm<false, false, true, false><<<gN512, blk, 0, stream>>>(
        ob, Wob + wE, bo + i * EMB, nullptr, t1f, nullptr, EMB, EMB);
    k_ln<true, true, true><<<NT / 8, blk, 0, stream>>>(
        t1f, hf, g1 + i * EMB, be1 + i * EMB, af, ab);
    k_gemm<false, true, false, true><<<gN2048, blk, 0, stream>>>(
        ab, W1b + (size_t)i * EMB * FFN, b1 + i * FFN, nullptr, nullptr, f1b,
        FFN, EMB);
    k_gemm<false, false, true, false><<<gN512, blk, 0, stream>>>(
        f1b, W2b + (size_t)i * FFN * EMB, b2 + i * EMB, nullptr, t1f, nullptr,
        EMB, FFN);
    k_ln<true, true, true><<<NT / 8, blk, 0, stream>>>(
        t1f, af, g2 + i * EMB, be2 + i * EMB, hf, hb);
  }

  // final LN + output head
  k_ln<false, true, false><<<NT / 8, blk, 0, stream>>>(
      hf, nullptr, gf, bef, af, nullptr);
  k_out<<<NT * 2 / 256, blk, 0, stream>>>(af, Wout, bout, (float*)d_out);
}
